// LlamaMlpWithLora_88313117541049
// MI455X (gfx1250) — matrix operand order repaired and checked by reading, not compiled, not run on hardware
//
#include <hip/hip_runtime.h>
#include <hip/hip_bf16.h>

#define T_TOK 2048
#define H_DIM 4096
#define I_DIM 11008
#define R_LORA 16
#define LSCALE 0.5f

typedef __bf16 v16bf __attribute__((ext_vector_type(16)));
typedef __bf16 v8bf  __attribute__((ext_vector_type(8)));
typedef float  v8f   __attribute__((ext_vector_type(8)));

__device__ __forceinline__ unsigned short f2bf(float f) {
  unsigned u = __float_as_uint(f);
  u += 0x7FFFu + ((u >> 16) & 1u);          // round-to-nearest-even
  return (unsigned short)(u >> 16);
}
__device__ __forceinline__ float bf2f(unsigned short s) {
  return __uint_as_float(((unsigned)s) << 16);
}

// CDNA5 async copy: per-lane 16B global -> LDS, tracked by ASYNCcnt.
__device__ __forceinline__ void async_copy_b128(void* lds_dst, const void* gsrc) {
  unsigned loff = (unsigned)(unsigned long long)lds_dst;   // low 32b of LDS-aperture addr
  unsigned long long ga = (unsigned long long)gsrc;
  asm volatile("global_load_async_to_lds_b128 %0, %1, off"
               :: "v"(loff), "v"(ga) : "memory");
}
// Wait until at most N async ops outstanding (async loads complete in order,
// so N == ops of the freshly-issued stage implies the previous stage landed).
#define ASYNC_WAIT(N) asm volatile("s_wait_asynccnt " #N ::: "memory")

// ---------------- fp32 -> bf16 conversion pre-pass ----------------
__global__ __launch_bounds__(256)
void cvt_f32_bf16(const float* __restrict__ src, unsigned short* __restrict__ dst,
                  long long n) {
  long long i = ((long long)blockIdx.x * 256 + threadIdx.x) * 4;
  if (i + 3 < n) {
    float4 v = *(const float4*)(src + i);
    ushort4 o;
    o.x = f2bf(v.x); o.y = f2bf(v.y); o.z = f2bf(v.z); o.w = f2bf(v.w);
    *(ushort4*)(dst + i) = o;
  }
}

// ---------------- per-token LoRA v = x . wa^T  (gate & up) ----------------
__global__ __launch_bounds__(256)
void v_gateup_kernel(const float* __restrict__ x, const float* __restrict__ gwa,
                     const float* __restrict__ uwa, const int* __restrict__ idx,
                     float* __restrict__ vg, float* __restrict__ vu) {
  __shared__ float xrow[H_DIM];          // 16 KB
  __shared__ float part[2][16][17];
  const int t = blockIdx.x, tid = threadIdx.x;
  for (int h = tid * 4; h < H_DIM; h += 1024)
    *(float4*)&xrow[h] = *(const float4*)&x[(size_t)t * H_DIM + h];
  __syncthreads();
  const int a = idx[t];
  const int r = tid >> 4, l = tid & 15;
  const float* g = gwa + ((size_t)a * R_LORA + r) * H_DIM;
  const float* u = uwa + ((size_t)a * R_LORA + r) * H_DIM;
  float sg = 0.f, su = 0.f;
  for (int h = l * 4; h < H_DIM; h += 64) {
    float4 xv = *(const float4*)&xrow[h];
    float4 gv = *(const float4*)&g[h];
    float4 uv = *(const float4*)&u[h];
    sg += xv.x * gv.x + xv.y * gv.y + xv.z * gv.z + xv.w * gv.w;
    su += xv.x * uv.x + xv.y * uv.y + xv.z * uv.z + xv.w * uv.w;
  }
  part[0][r][l] = sg; part[1][r][l] = su;
  __syncthreads();
  if (tid < 16) {
    float s0 = 0.f, s1 = 0.f;
    for (int j = 0; j < 16; ++j) { s0 += part[0][tid][j]; s1 += part[1][tid][j]; }
    vg[t * R_LORA + tid] = s0;
    vu[t * R_LORA + tid] = s1;
  }
}

// ---------------- per-token LoRA v_down = t . down_wa^T ----------------
__global__ __launch_bounds__(256)
void v_down_kernel(const unsigned short* __restrict__ tb, const float* __restrict__ dwa,
                   const int* __restrict__ idx, float* __restrict__ vd) {
  __shared__ float trow[I_DIM];          // 44 KB
  __shared__ float part[16][17];
  const int t = blockIdx.x, tid = threadIdx.x;
  for (int i = tid * 4; i < I_DIM; i += 1024) {
    ushort4 q = *(const ushort4*)&tb[(size_t)t * I_DIM + i];
    trow[i] = bf2f(q.x); trow[i + 1] = bf2f(q.y);
    trow[i + 2] = bf2f(q.z); trow[i + 3] = bf2f(q.w);
  }
  __syncthreads();
  const int a = idx[t];
  const int r = tid >> 4, l = tid & 15;
  const float* w = dwa + ((size_t)a * R_LORA + r) * I_DIM;
  float s = 0.f;
  for (int i = l * 4; i < I_DIM; i += 64) {
    float4 wv = *(const float4*)&w[i];
    s += trow[i] * wv.x + trow[i + 1] * wv.y + trow[i + 2] * wv.z + trow[i + 3] * wv.w;
  }
  part[r][l] = s;
  __syncthreads();
  if (tid < 16) {
    float s0 = 0.f;
    for (int j = 0; j < 16; ++j) s0 += part[tid][j];
    vd[t * R_LORA + tid] = s0;
  }
}

// ============================ gate/up GEMM ============================
// 256 thr = 8 waves as 2(waveM) x 4(waveN); wave tile 32M x 64N (2x4 subtiles);
// block tile 64M x 256N; K-stage 64 (two 32-K sub-steps); double-buffered
// async-LDS staging; steady-state loop body is branch-free.
// Stage LDS layout: [sub-k(2)][row/col][32] so each 32-K sub-step is contiguous.

__device__ __forceinline__ void issue_gateup(const unsigned short* __restrict__ arow,
                                             const unsigned short* __restrict__ gs0,
                                             const unsigned short* __restrict__ us0,
                                             int k0, unsigned short* aB,
                                             unsigned short* gB, unsigned short* uB,
                                             int tid) {
  const int row = tid >> 2, c4 = (tid & 3) * 8;            // 64 rows x 4 chunks/sub-k
  const unsigned short* as = arow + (size_t)row * H_DIM + k0 + c4;
#pragma unroll
  for (int p = 0; p < 2; ++p)                              // 2 A ops: sub-k 0,1
    async_copy_b128(&aB[p * 2048 + row * 32 + c4], as + p * 32);
  const unsigned short* gs = gs0 + k0;
  const unsigned short* us = us0 + k0;
#pragma unroll
  for (int p = 0; p < 8; ++p) {                            // 8+8 B ops
    const int d = (p >> 2) * 8192 + tid * 32 + (p & 3) * 8;
    async_copy_b128(&gB[d], gs + p * 8);
    async_copy_b128(&uB[d], us + p * 8);
  }
}

__global__ __launch_bounds__(256)
void gateup_gemm(const unsigned short* __restrict__ xb,
                 const unsigned short* __restrict__ gwb,
                 const unsigned short* __restrict__ uwb,
                 const float* __restrict__ gate_wb,
                 const float* __restrict__ up_wb,
                 const float* __restrict__ vg,
                 const float* __restrict__ vu,
                 const int* __restrict__ idx,
                 unsigned short* __restrict__ tb) {
  __shared__ unsigned short aS[2][2 * 64 * 32];    // 2 x 8 KB
  __shared__ unsigned short bSg[2][2 * 256 * 32];  // 2 x 32 KB
  __shared__ unsigned short bSu[2][2 * 256 * 32];  // 2 x 32 KB

  const int tid = threadIdx.x;
  const int tok0 = blockIdx.x * 64;               // M fastest -> L2 reuse of B tiles
  const int col0 = blockIdx.y * 256;

  const int wave = tid >> 5, lane = tid & 31;
  const int m0 = (wave & 1) * 32;
  const int n0 = (wave >> 1) * 64;
  const int lrow = lane & 15, lhalf = lane >> 4;

  const v8f zero = {0.f, 0.f, 0.f, 0.f, 0.f, 0.f, 0.f, 0.f};
  v8f accg[2][4], accu[2][4];
#pragma unroll
  for (int mi = 0; mi < 2; ++mi)
#pragma unroll
    for (int s = 0; s < 4; ++s) { accg[mi][s] = zero; accu[mi][s] = zero; }

  const unsigned short* arow = xb + (size_t)tok0 * H_DIM;
  const unsigned short* gs0 = gwb + (size_t)(col0 + tid) * H_DIM;
  const unsigned short* us0 = uwb + (size_t)(col0 + tid) * H_DIM;

  auto compute = [&](int cb) {
#pragma unroll
    for (int ks = 0; ks < 2; ++ks) {
      const unsigned short* aP = &aS[cb][ks * 2048];
      const unsigned short* gP = &bSg[cb][ks * 8192];
      const unsigned short* uP = &bSu[cb][ks * 8192];
      union AF { v16bf v; v8bf h[2]; } af[2];
#pragma unroll
      for (int mi = 0; mi < 2; ++mi) {
        const int row = m0 + mi * 16 + lrow;
        af[mi].h[0] = *(const v8bf*)&aP[row * 32 + lhalf * 16];
        af[mi].h[1] = *(const v8bf*)&aP[row * 32 + lhalf * 16 + 8];
      }
#pragma unroll
      for (int s = 0; s < 4; ++s) {
        const int bo = (n0 + s * 16 + lrow) * 32 + lhalf * 16;
        v16bf bg = *(const v16bf*)&gP[bo];
        accg[0][s] = __builtin_amdgcn_wmma_f32_16x16x32_bf16(
            false, af[0].v, false, bg, (short)0, accg[0][s], false, false);
        accg[1][s] = __builtin_amdgcn_wmma_f32_16x16x32_bf16(
            false, af[1].v, false, bg, (short)0, accg[1][s], false, false);
        v16bf bu = *(const v16bf*)&uP[bo];
        accu[0][s] = __builtin_amdgcn_wmma_f32_16x16x32_bf16(
            false, af[0].v, false, bu, (short)0, accu[0][s], false, false);
        accu[1][s] = __builtin_amdgcn_wmma_f32_16x16x32_bf16(
            false, af[1].v, false, bu, (short)0, accu[1][s], false, false);
      }
    }
  };

  issue_gateup(arow, gs0, us0, 0, aS[0], bSg[0], bSu[0], tid);
  int cur = 0;
  for (int k0 = 0; k0 + 64 < H_DIM; k0 += 64) {   // steady state: branch-free body
    issue_gateup(arow, gs0, us0, k0 + 64, aS[cur ^ 1], bSg[cur ^ 1], bSu[cur ^ 1], tid);
    ASYNC_WAIT(18);           // 18 new ops outstanding => previous stage landed
    __syncthreads();
    compute(cur);
    __syncthreads();          // everyone done reading 'cur' before it is refilled
    cur ^= 1;
  }
  ASYNC_WAIT(0);              // peeled final stage
  __syncthreads();
  compute(cur);

  // Epilogue: C layout -> VGPR j : M = j + 8*(lane>>4), N = lane&15
#pragma unroll
  for (int mi = 0; mi < 2; ++mi) {
#pragma unroll
    for (int s = 0; s < 4; ++s) {
      const int ccol = col0 + n0 + s * 16 + (lane & 15);
#pragma unroll
      for (int j = 0; j < 8; ++j) {
        const int tok = tok0 + m0 + mi * 16 + (lane >> 4) * 8 + j;
        const int a = idx[tok];
        const float4* wbg = (const float4*)(gate_wb + ((size_t)a * I_DIM + ccol) * R_LORA);
        const float4* wbu = (const float4*)(up_wb + ((size_t)a * I_DIM + ccol) * R_LORA);
        const float4* pg = (const float4*)(vg + tok * R_LORA);
        const float4* pu = (const float4*)(vu + tok * R_LORA);
        float lg = 0.f, lu = 0.f;
#pragma unroll
        for (int q = 0; q < 4; ++q) {
          float4 wg = wbg[q], wu = wbu[q], xg = pg[q], xu = pu[q];
          lg += xg.x * wg.x + xg.y * wg.y + xg.z * wg.z + xg.w * wg.w;
          lu += xu.x * wu.x + xu.y * wu.y + xu.z * wu.z + xu.w * wu.w;
        }
        float g = accg[mi][s][j] + LSCALE * lg;
        float u = accu[mi][s][j] + LSCALE * lu;
        float sg = g / (1.f + __expf(-g));      // silu
        tb[(size_t)tok * I_DIM + ccol] = f2bf(sg * u);
      }
    }
  }
}

// ============================ down GEMM ============================
__device__ __forceinline__ void issue_down(const unsigned short* __restrict__ arow,
                                           const unsigned short* __restrict__ bs0,
                                           int k0, unsigned short* aB,
                                           unsigned short* bB, int tid) {
  const int row = tid >> 2, c4 = (tid & 3) * 8;
  const unsigned short* as = arow + (size_t)row * I_DIM + k0 + c4;
#pragma unroll
  for (int p = 0; p < 2; ++p)
    async_copy_b128(&aB[p * 2048 + row * 32 + c4], as + p * 32);
  const unsigned short* bs = bs0 + k0;
#pragma unroll
  for (int p = 0; p < 8; ++p)
    async_copy_b128(&bB[(p >> 2) * 8192 + tid * 32 + (p & 3) * 8], bs + p * 8);
}

__global__ __launch_bounds__(256)
void down_gemm(const unsigned short* __restrict__ tbA,
               const unsigned short* __restrict__ dwb_w,
               const float* __restrict__ down_wb,
               const float* __restrict__ vd,
               const int* __restrict__ idx,
               float* __restrict__ out) {
  __shared__ unsigned short aS[2][2 * 64 * 32];    // 2 x 8 KB
  __shared__ unsigned short bS[2][2 * 256 * 32];   // 2 x 32 KB

  const int tid = threadIdx.x;
  const int tok0 = blockIdx.x * 64;
  const int col0 = blockIdx.y * 256;

  const int wave = tid >> 5, lane = tid & 31;
  const int m0 = (wave & 1) * 32;
  const int n0 = (wave >> 1) * 64;
  const int lrow = lane & 15, lhalf = lane >> 4;

  const v8f zero = {0.f, 0.f, 0.f, 0.f, 0.f, 0.f, 0.f, 0.f};
  v8f acc[2][4];
#pragma unroll
  for (int mi = 0; mi < 2; ++mi)
#pragma unroll
    for (int s = 0; s < 4; ++s) acc[mi][s] = zero;

  const unsigned short* arow = tbA + (size_t)tok0 * I_DIM;
  const unsigned short* bs0 = dwb_w + (size_t)(col0 + tid) * I_DIM;

  auto compute = [&](int cb) {
#pragma unroll
    for (int ks = 0; ks < 2; ++ks) {
      const unsigned short* aP = &aS[cb][ks * 2048];
      const unsigned short* bP = &bS[cb][ks * 8192];
      union AF { v16bf v; v8bf h[2]; } af[2];
#pragma unroll
      for (int mi = 0; mi < 2; ++mi) {
        const int row = m0 + mi * 16 + lrow;
        af[mi].h[0] = *(const v8bf*)&aP[row * 32 + lhalf * 16];
        af[mi].h[1] = *(const v8bf*)&aP[row * 32 + lhalf * 16 + 8];
      }
#pragma unroll
      for (int s = 0; s < 4; ++s) {
        v16bf b = *(const v16bf*)&bP[(n0 + s * 16 + lrow) * 32 + lhalf * 16];
        acc[0][s] = __builtin_amdgcn_wmma_f32_16x16x32_bf16(
            false, af[0].v, false, b, (short)0, acc[0][s], false, false);
        acc[1][s] = __builtin_amdgcn_wmma_f32_16x16x32_bf16(
            false, af[1].v, false, b, (short)0, acc[1][s], false, false);
      }
    }
  };

  issue_down(arow, bs0, 0, aS[0], bS[0], tid);
  int cur = 0;
  for (int k0 = 0; k0 + 64 < I_DIM; k0 += 64) {
    issue_down(arow, bs0, k0 + 64, aS[cur ^ 1], bS[cur ^ 1], tid);
    ASYNC_WAIT(10);           // 10 new ops outstanding => previous stage landed
    __syncthreads();
    compute(cur);
    __syncthreads();
    cur ^= 1;
  }
  ASYNC_WAIT(0);
  __syncthreads();
  compute(cur);

#pragma unroll
  for (int mi = 0; mi < 2; ++mi) {
#pragma unroll
    for (int s = 0; s < 4; ++s) {
      const int ccol = col0 + n0 + s * 16 + (lane & 15);
#pragma unroll
      for (int j = 0; j < 8; ++j) {
        const int tok = tok0 + m0 + mi * 16 + (lane >> 4) * 8 + j;
        const int a = idx[tok];
        const float4* wb = (const float4*)(down_wb + ((size_t)a * H_DIM + ccol) * R_LORA);
        const float4* pv = (const float4*)(vd + tok * R_LORA);
        float lo = 0.f;
#pragma unroll
        for (int q = 0; q < 4; ++q) {
          float4 w = wb[q], v = pv[q];
          lo += v.x * w.x + v.y * w.y + v.z * w.z + v.w * w.w;
        }
        out[(size_t)tok * H_DIM + ccol] = acc[mi][s][j] + LSCALE * lo;
      }
    }
  }
}

extern "C" void kernel_launch(void* const* d_in, const int* in_sizes, int n_in,
                              void* d_out, int out_size, void* d_ws, size_t ws_size,
                              hipStream_t stream) {
  const float* x       = (const float*)d_in[0];
  const float* gate_w  = (const float*)d_in[1];
  const float* up_w    = (const float*)d_in[2];
  const float* down_w  = (const float*)d_in[3];
  const float* gate_wa = (const float*)d_in[4];
  const float* gate_wb = (const float*)d_in[5];
  const float* up_wa   = (const float*)d_in[6];
  const float* up_wb   = (const float*)d_in[7];
  const float* down_wa = (const float*)d_in[8];
  const float* down_wb = (const float*)d_in[9];
  const int*   indices = (const int*)d_in[10];
  float* out = (float*)d_out;
  (void)in_sizes; (void)n_in; (void)out_size; (void)ws_size;

  // Workspace layout (bf16 copies + t + LoRA v vectors), ~332 MB total.
  unsigned short* xb  = (unsigned short*)d_ws;
  unsigned short* gwb = xb + (size_t)T_TOK * H_DIM;
  unsigned short* uwb = gwb + (size_t)I_DIM * H_DIM;
  unsigned short* dwb = uwb + (size_t)I_DIM * H_DIM;
  unsigned short* tb  = dwb + (size_t)H_DIM * I_DIM;
  float* vg = (float*)(tb + (size_t)T_TOK * I_DIM);
  float* vu = vg + (size_t)T_TOK * R_LORA;
  float* vd = vu + (size_t)T_TOK * R_LORA;

  {
    long long n;
    n = (long long)T_TOK * H_DIM;
    cvt_f32_bf16<<<(unsigned)((n / 4 + 255) / 256), 256, 0, stream>>>(x, xb, n);
    n = (long long)I_DIM * H_DIM;
    cvt_f32_bf16<<<(unsigned)((n / 4 + 255) / 256), 256, 0, stream>>>(gate_w, gwb, n);
    cvt_f32_bf16<<<(unsigned)((n / 4 + 255) / 256), 256, 0, stream>>>(up_w, uwb, n);
    n = (long long)H_DIM * I_DIM;
    cvt_f32_bf16<<<(unsigned)((n / 4 + 255) / 256), 256, 0, stream>>>(down_w, dwb, n);
  }

  v_gateup_kernel<<<T_TOK, 256, 0, stream>>>(x, gate_wa, up_wa, indices, vg, vu);

  gateup_gemm<<<dim3(T_TOK / 64, I_DIM / 256), 256, 0, stream>>>(
      xb, gwb, uwb, gate_wb, up_wb, vg, vu, indices, tb);

  v_down_kernel<<<T_TOK, 256, 0, stream>>>(tb, down_wa, indices, vd);

  down_gemm<<<dim3(T_TOK / 64, H_DIM / 256), 256, 0, stream>>>(
      tb, dwb, down_wb, vd, indices, out);
}
